// StrategicPlanningLayer_47004122088029
// MI455X (gfx1250) — matrix operand order repaired and checked by reading, not compiled, not run on hardware
//
#include <hip/hip_runtime.h>
#include <hip/hip_bf16.h>

// ---------------- Types ----------------
typedef __attribute__((ext_vector_type(16))) __bf16 v16bf;
typedef __attribute__((ext_vector_type(8)))  float  v8f;
typedef __attribute__((ext_vector_type(4)))  unsigned int v4u;
typedef __attribute__((ext_vector_type(4)))  int v4i;

#define POP   4096
#define DDIM  512
#define GENS  50

// ---------------- CDNA5 async global->LDS (probe-guarded) ----------------
#if defined(__has_builtin)
#  if __has_builtin(__builtin_amdgcn_global_load_async_to_lds_b128)
#    define HAVE_ASYNC_G2L 1
#  endif
#  if __has_builtin(__builtin_amdgcn_s_wait_asynccnt)
#    define HAVE_WAIT_ASYNC 1
#  endif
#endif
#ifndef HAVE_ASYNC_G2L
#  define HAVE_ASYNC_G2L 0
#endif
#ifndef HAVE_WAIT_ASYNC
#  define HAVE_WAIT_ASYNC 0
#endif

typedef __attribute__((address_space(1))) v4i gv4i;   // global int4
typedef __attribute__((address_space(3))) v4i lv4i;   // LDS int4

__device__ __forceinline__ void g2l_b128(const __bf16* g, __bf16* l) {
#if HAVE_ASYNC_G2L
    // GLOBAL_LOAD_ASYNC_TO_LDS_B128: memory -> LDS without VGPR round-trip (ASYNCcnt)
    __builtin_amdgcn_global_load_async_to_lds_b128((gv4i*)g, (lv4i*)l, 0, 0);
#else
    *(v4u*)l = *(const v4u*)g;
#endif
}
__device__ __forceinline__ void wait_g2l() {
#if HAVE_ASYNC_G2L && HAVE_WAIT_ASYNC
    __builtin_amdgcn_s_wait_asynccnt(0);
#endif
}

// ---------------- RNG helpers (counter-based; stands in for JAX PRNG) ----------------
__device__ __forceinline__ unsigned pcg_hash(unsigned x) {
    x = x * 747796405u + 2891336453u;
    unsigned w = ((x >> ((x >> 28) + 4u)) ^ x) * 277803737u;
    return (w >> 22) ^ w;
}
__device__ __forceinline__ unsigned hash2(unsigned a, unsigned b) {
    return pcg_hash(a * 0x9E3779B9u + b + 0x7F4A7C15u);
}
__device__ __forceinline__ float u01f(unsigned s) {
    return (float)(pcg_hash(s) >> 8) * 5.9604645e-8f;
}
__device__ __forceinline__ float nrmf(unsigned s) {
    float u1 = fmaxf(u01f(s), 1e-7f);
    float u2 = u01f(s ^ 0x85EBCA6Bu);
    return sqrtf(-2.0f * __logf(u1)) * __cosf(6.2831853f * u2);
}

// ---------------- 1) Encoder (S=1 -> attention output == v) ----------------
__global__ __launch_bounds__(512) void enc_kernel(
    const float* __restrict__ pc,
    const float* __restrict__ qkv_w, const float* __restrict__ qkv_b,
    const float* __restrict__ out_w, const float* __restrict__ out_b,
    const float* __restrict__ ff1_w, const float* __restrict__ ff1_b,
    const float* __restrict__ ff2_w, const float* __restrict__ ff2_b,
    const float* __restrict__ ln1_w, const float* __restrict__ ln1_b,
    const float* __restrict__ ln2_w, const float* __restrict__ ln2_b,
    float* __restrict__ ctx)
{
    __shared__ float xs[512], vs[512], t[512], red[512];
    __shared__ float hf[2048];
    const int tid = threadIdx.x;
    xs[tid] = pc[tid];
    __syncthreads();

    for (int L = 0; L < 6; ++L) {
        {
            const float* wr = qkv_w + ((size_t)L * 1536 + 1024 + tid) * 512;
            float s = qkv_b[L * 1536 + 1024 + tid];
            for (int k = 0; k < 512; ++k) s += wr[k] * xs[k];
            vs[tid] = s;
        }
        __syncthreads();
        {
            const float* wr = out_w + ((size_t)L * 512 + tid) * 512;
            float s = out_b[L * 512 + tid];
            for (int k = 0; k < 512; ++k) s += wr[k] * vs[k];
            t[tid] = xs[tid] + s;
        }
        __syncthreads();
        {   // LN1: t -> xs
            red[tid] = t[tid]; __syncthreads();
            for (int s = 256; s > 0; s >>= 1) { if (tid < s) red[tid] += red[tid + s]; __syncthreads(); }
            float mean = red[0] * (1.0f / 512.0f); __syncthreads();
            float dv = t[tid] - mean; red[tid] = dv * dv; __syncthreads();
            for (int s = 256; s > 0; s >>= 1) { if (tid < s) red[tid] += red[tid + s]; __syncthreads(); }
            float var = red[0] * (1.0f / 512.0f); __syncthreads();
            xs[tid] = dv * rsqrtf(var + 1e-5f) * ln1_w[L * 512 + tid] + ln1_b[L * 512 + tid];
        }
        __syncthreads();
        for (int o = tid; o < 2048; o += 512) {
            const float* wr = ff1_w + ((size_t)L * 2048 + o) * 512;
            float s = ff1_b[L * 2048 + o];
            for (int k = 0; k < 512; ++k) s += wr[k] * xs[k];
            hf[o] = fmaxf(s, 0.0f);
        }
        __syncthreads();
        {
            const float* wr = ff2_w + ((size_t)L * 512 + tid) * 2048;
            float s = ff2_b[L * 512 + tid];
            for (int k = 0; k < 2048; ++k) s += wr[k] * hf[k];
            t[tid] = xs[tid] + s;
        }
        __syncthreads();
        {   // LN2: t -> xs
            red[tid] = t[tid]; __syncthreads();
            for (int s = 256; s > 0; s >>= 1) { if (tid < s) red[tid] += red[tid + s]; __syncthreads(); }
            float mean = red[0] * (1.0f / 512.0f); __syncthreads();
            float dv = t[tid] - mean; red[tid] = dv * dv; __syncthreads();
            for (int s = 256; s > 0; s >>= 1) { if (tid < s) red[tid] += red[tid + s]; __syncthreads(); }
            float var = red[0] * (1.0f / 512.0f); __syncthreads();
            xs[tid] = dv * rsqrtf(var + 1e-5f) * ln2_w[L * 512 + tid] + ln2_b[L * 512 + tid];
        }
        __syncthreads();
    }
    ctx[tid] = xs[tid];
}

// ---------------- 2) Row softmax of entanglement, cast to bf16 (once; 33.5MB -> L2-resident) ----
__global__ __launch_bounds__(256) void softmax_bf16_kernel(
    const float* __restrict__ raw, __bf16* __restrict__ entb)
{
    __shared__ float red[256];
    const int row = blockIdx.x, tid = threadIdx.x;
    const float* r = raw + (size_t)row * POP;
    float mx = -3.4e38f;
    for (int c = tid; c < POP; c += 256) mx = fmaxf(mx, r[c]);
    red[tid] = mx; __syncthreads();
    for (int s = 128; s > 0; s >>= 1) { if (tid < s) red[tid] = fmaxf(red[tid], red[tid + s]); __syncthreads(); }
    mx = red[0]; __syncthreads();
    float sm = 0.0f;
    for (int c = tid; c < POP; c += 256) sm += __expf(r[c] - mx);
    red[tid] = sm; __syncthreads();
    for (int s = 128; s > 0; s >>= 1) { if (tid < s) red[tid] += red[tid + s]; __syncthreads(); }
    const float inv = 1.0f / red[0];
    for (int c = tid; c < POP; c += 256)
        entb[(size_t)row * POP + c] = (__bf16)(__expf(r[c] - mx) * inv);
}

// ---------------- 3) Population init ----------------
__global__ __launch_bounds__(256) void initpop_kernel(
    const float* __restrict__ ctx, float* __restrict__ pop)
{
    const int gid = blockIdx.x * 256 + threadIdx.x;
    const int n = gid & (DDIM - 1);
    pop[gid] = 0.1f * ctx[n] + nrmf(hash2(0x1234567u, (unsigned)gid));
}

// ---------------- 4a) combinedT = (pop + ctx)^T in bf16 ([512][4096]) ----------------
__global__ __launch_bounds__(256) void combt_kernel(
    const float* __restrict__ pop, const float* __restrict__ ctx,
    __bf16* __restrict__ combT)
{
    const int gid = blockIdx.x * 256 + threadIdx.x;   // 0 .. 512*4096-1
    const int n = gid >> 12;            // / 4096
    const int m = gid & (POP - 1);
    combT[(size_t)n * POP + m] = (__bf16)(pop[(size_t)m * DDIM + n] + ctx[n]);
}

// ---------------- 4b) Main bf16 WMMA GEMM: qe = combined + 0.3 * (ent @ combined) ----------------
// Block tile 128x128, 8 waves of 32x64, K steps of 32, double-buffered LDS,
// async global->LDS staging (ASYNCcnt) when available.
#define LDSROW 40   // 32 + 8 pad elems -> 80B row stride (16B-aligned rows)

__device__ __forceinline__ v16bf frag_a(const __bf16* p) {
    // 16-bit A 16x32 layout: elems 0..7 = K(kk+0..7), elems 8..15 = K(kk+16..23)
    union { v16bf v; v4u u[2]; } r;
    r.u[0] = *(const v4u*)(p);
    r.u[1] = *(const v4u*)(p + 16);
    return r.v;
}
__device__ __forceinline__ v16bf frag_b(const __bf16* p) {
    // B 32x16: lane holds column, elems 0..7 = K(kk+0..7), elems 8..15 = K(kk+16..23)
    union { v16bf v; v4u u[2]; } r;
    r.u[0] = *(const v4u*)(p);
    r.u[1] = *(const v4u*)(p + 16);
    return r.v;
}

__global__ __launch_bounds__(256) void gemm_qe_kernel(
    const __bf16* __restrict__ entb,   // [4096][4096]
    const __bf16* __restrict__ combT,  // [512][4096]
    const float*  __restrict__ pop,    // [4096][512]
    const float*  __restrict__ ctx,    // [512]
    float* __restrict__ qe)            // [4096][512]
{
    __shared__ __bf16 As[2][128][LDSROW];
    __shared__ __bf16 Bs[2][128][LDSROW];

    const int tid  = threadIdx.x;
    const int wave = tid >> 5;
    const int lane = tid & 31;
    const int m_blk = blockIdx.y * 128;
    const int n_blk = blockIdx.x * 128;
    const int wm = wave >> 1;      // 0..3  (M)
    const int wn = wave & 1;       // 0..1  (N)

    v8f acc[2][4];
    const v8f zero = {0, 0, 0, 0, 0, 0, 0, 0};
    for (int i = 0; i < 2; ++i)
        for (int j = 0; j < 4; ++j) acc[i][j] = zero;

    // staging map: 256 threads cover a full 128x32 bf16 tile: 16 bf16 per thread
    const int crow  = tid >> 1;          // 0..127
    const int chalf = (tid & 1) * 16;    // element offset 0 or 16 within 32-wide slab

    const size_t gA = (size_t)(m_blk + crow) * POP + chalf;
    const size_t gB = (size_t)(n_blk + crow) * POP + chalf;

    // stage k-tile 0 (both b128 chunks -> full coverage)
    g2l_b128(entb  + gA,     &As[0][crow][chalf]);
    g2l_b128(entb  + gA + 8, &As[0][crow][chalf + 8]);
    g2l_b128(combT + gB,     &Bs[0][crow][chalf]);
    g2l_b128(combT + gB + 8, &Bs[0][crow][chalf + 8]);
    wait_g2l();
    __syncthreads();

    const int KT = POP / 32;   // 128
    for (int kt = 0; kt < KT; ++kt) {
        const int cur = kt & 1, nxt = cur ^ 1;
        if (kt + 1 < KT) {
            const size_t ko = (size_t)(kt + 1) * 32;
            g2l_b128(entb  + gA + ko,     &As[nxt][crow][chalf]);
            g2l_b128(entb  + gA + ko + 8, &As[nxt][crow][chalf + 8]);
            g2l_b128(combT + gB + ko,     &Bs[nxt][crow][chalf]);
            g2l_b128(combT + gB + ko + 8, &Bs[nxt][crow][chalf + 8]);
        }
        if (kt + 2 < KT) {
            __builtin_prefetch(entb  + gA + (size_t)(kt + 2) * 32, 0, 0);
            __builtin_prefetch(combT + gB + (size_t)(kt + 2) * 32, 0, 0);
        }

        const int arow = wm * 32 + (lane & 15);
        const int akk  = (lane >> 4) * 8;
        v16bf a0 = frag_a(&As[cur][arow     ][akk]);
        v16bf a1 = frag_a(&As[cur][arow + 16][akk]);

        const int brow = wn * 64 + (lane & 15);
        const int bkk  = (lane >> 4) * 8;
        v16bf b0 = frag_b(&Bs[cur][brow     ][bkk]);
        v16bf b1 = frag_b(&Bs[cur][brow + 16][bkk]);
        v16bf b2 = frag_b(&Bs[cur][brow + 32][bkk]);
        v16bf b3 = frag_b(&Bs[cur][brow + 48][bkk]);

        acc[0][0] = __builtin_amdgcn_wmma_f32_16x16x32_bf16(false, a0, false, b0, (short)0, acc[0][0], false, false);
        acc[0][1] = __builtin_amdgcn_wmma_f32_16x16x32_bf16(false, a0, false, b1, (short)0, acc[0][1], false, false);
        acc[0][2] = __builtin_amdgcn_wmma_f32_16x16x32_bf16(false, a0, false, b2, (short)0, acc[0][2], false, false);
        acc[0][3] = __builtin_amdgcn_wmma_f32_16x16x32_bf16(false, a0, false, b3, (short)0, acc[0][3], false, false);
        acc[1][0] = __builtin_amdgcn_wmma_f32_16x16x32_bf16(false, a1, false, b0, (short)0, acc[1][0], false, false);
        acc[1][1] = __builtin_amdgcn_wmma_f32_16x16x32_bf16(false, a1, false, b1, (short)0, acc[1][1], false, false);
        acc[1][2] = __builtin_amdgcn_wmma_f32_16x16x32_bf16(false, a1, false, b2, (short)0, acc[1][2], false, false);
        acc[1][3] = __builtin_amdgcn_wmma_f32_16x16x32_bf16(false, a1, false, b3, (short)0, acc[1][3], false, false);

        wait_g2l();
        __syncthreads();
    }

    // Epilogue: C/D layout -> (M = r + (lane>>4)*8, N = lane&15) per 16x16 frag
    const int mo = m_blk + wm * 32 + (lane >> 4) * 8;
    const int no = n_blk + wn * 64 + (lane & 15);
    for (int mf = 0; mf < 2; ++mf) {
        for (int nf = 0; nf < 4; ++nf) {
            const int n = no + nf * 16;
            const float cx = ctx[n];
            for (int r = 0; r < 8; ++r) {
                const int m = mo + mf * 16 + r;
                const size_t idx = (size_t)m * DDIM + n;
                qe[idx] = pop[idx] + cx + 0.3f * acc[mf][nf][r];
            }
        }
    }
}

// ---------------- 4c) Fitness MLP: sigmoid(relu(relu(qe@W1)@W2)@W3) + noise ----------------
__global__ __launch_bounds__(256) void fitness_kernel(
    const float* __restrict__ qe,
    const float* __restrict__ w1, const float* __restrict__ b1,
    const float* __restrict__ w2, const float* __restrict__ b2,
    const float* __restrict__ w3, const float* __restrict__ b3,
    float* __restrict__ fit, int gen)
{
    __shared__ float xs[512], h1[256], h2[128], red[256];
    const int row = blockIdx.x, tid = threadIdx.x;
    xs[tid]       = qe[(size_t)row * DDIM + tid];
    xs[tid + 256] = qe[(size_t)row * DDIM + 256 + tid];
    __syncthreads();
    {
        const float* wr = w1 + (size_t)tid * 512;
        float s = b1[tid];
        for (int k = 0; k < 512; ++k) s += wr[k] * xs[k];
        h1[tid] = fmaxf(s, 0.0f);
    }
    __syncthreads();
    if (tid < 128) {
        const float* wr = w2 + (size_t)tid * 256;
        float s = b2[tid];
        for (int k = 0; k < 256; ++k) s += wr[k] * h1[k];
        h2[tid] = fmaxf(s, 0.0f);
    }
    __syncthreads();
    red[tid] = (tid < 128) ? h2[tid] * w3[tid] : 0.0f;
    __syncthreads();
    for (int s = 128; s > 0; s >>= 1) { if (tid < s) red[tid] += red[tid + s]; __syncthreads(); }
    if (tid == 0) {
        const float z = red[0] + b3[0];
        const float sig = 1.0f / (1.0f + __expf(-z));
        fit[row] = sig + 0.1f * nrmf(hash2((unsigned)(gen + 1) * 7u, (unsigned)row));
    }
}

// ---------------- 4d) Tournament selection + crossover + mutation ----------------
__global__ __launch_bounds__(128) void reproduce_kernel(
    const float* __restrict__ pop, const float* __restrict__ fit,
    float* __restrict__ popn, int gen)
{
    __shared__ int pidx[2];
    const int pair = blockIdx.x, tid = threadIdx.x;
    if (tid < 2) {
        const int row = pair * 2 + tid;
        const unsigned base = hash2((unsigned)(gen + 1) * 13u + 5u, (unsigned)row);
        float best = -3.4e38f; int bi = 0;
        for (int t = 0; t < 5; ++t) {
            const unsigned hh = hash2(base, (unsigned)t);
            const int idx = (int)(hh & (POP - 1));
            const float g = -__logf(fmaxf(-__logf(fmaxf(u01f(hh ^ 0xDEADBEEFu), 1e-12f)), 1e-12f));
            const float sc = fit[idx] * 10.0f + g;   // Gumbel-max categorical
            if (sc > best) { best = sc; bi = idx; }
        }
        pidx[tid] = bi;
    }
    __syncthreads();
    const float* p1 = pop + (size_t)pidx[0] * DDIM;
    const float* p2 = pop + (size_t)pidx[1] * DDIM;
    const bool meas_on = ((gen % 10) == 0);
    for (int d = tid; d < DDIM; d += 128) {
        const unsigned sd = hash2((unsigned)(gen + 1) * 31u + 11u, (unsigned)(pair * DDIM + d));
        const float a = p1[d], b = p2[d];
        const bool  cm   = u01f(sd) < 0.5f;
        const float ef   = u01f(sd ^ 0x1111u) * 0.2f;
        const float mean = 0.5f * (a + b);
        float c1 = (cm ? a : b) + ef * mean;
        float c2 = (cm ? b : a) + ef * mean;
        const unsigned s1 = hash2(sd, 101u), s2 = hash2(sd, 202u);
        if (u01f(s1) < 0.15f)          c1 += nrmf(s1 ^ 0x5u) * 0.1f;
        if (u01f(s2) < 0.15f)          c2 += nrmf(s2 ^ 0x5u) * 0.1f;
        if (u01f(s1 ^ 0x77u) < 0.05f)  c1 += nrmf(s1 ^ 0x9u) * 0.5f;
        if (u01f(s2 ^ 0x77u) < 0.05f)  c2 += nrmf(s2 ^ 0x9u) * 0.5f;
        if (meas_on) {
            c1 += nrmf(s1 ^ 0x33u) * 0.05f;
            c2 += nrmf(s2 ^ 0x33u) * 0.05f;
        }
        popn[(size_t)(pair * 2 + 0) * DDIM + d] = c1;
        popn[(size_t)(pair * 2 + 1) * DDIM + d] = c2;
    }
}

// ---------------- 5) argmax(fit) + output assembly ----------------
__global__ __launch_bounds__(512) void finalize_kernel(
    const float* __restrict__ pop, const float* __restrict__ fit,
    const float* __restrict__ objw, float* __restrict__ out)
{
    __shared__ float bv[512]; __shared__ int bix[512];
    const int tid = threadIdx.x;
    float v = -3.4e38f; int bi = 0;
    for (int i = tid; i < POP; i += 512) { const float f = fit[i]; if (f > v) { v = f; bi = i; } }
    bv[tid] = v; bix[tid] = bi; __syncthreads();
    for (int s = 256; s > 0; s >>= 1) {
        if (tid < s && bv[tid + s] > bv[tid]) { bv[tid] = bv[tid + s]; bix[tid] = bix[tid + s]; }
        __syncthreads();
    }
    const int best = bix[0];
    out[tid] = pop[(size_t)best * DDIM + tid];
    if (tid == 0) out[DDIM] = bv[0];
    if (tid < 5)  out[DDIM + 1 + tid] = objw[tid];
}

// ---------------- Host launch ----------------
extern "C" void kernel_launch(void* const* d_in, const int* in_sizes, int n_in,
                              void* d_out, int out_size, void* d_ws, size_t ws_size,
                              hipStream_t stream)
{
    (void)in_sizes; (void)n_in; (void)out_size; (void)ws_size;
    const float* pc     = (const float*)d_in[0];
    const float* qkv_w  = (const float*)d_in[1];
    const float* qkv_b  = (const float*)d_in[2];
    const float* out_w  = (const float*)d_in[3];
    const float* out_b  = (const float*)d_in[4];
    const float* ff1_w  = (const float*)d_in[5];
    const float* ff1_b  = (const float*)d_in[6];
    const float* ff2_w  = (const float*)d_in[7];
    const float* ff2_b  = (const float*)d_in[8];
    const float* ln1_w  = (const float*)d_in[9];
    const float* ln1_b  = (const float*)d_in[10];
    const float* ln2_w  = (const float*)d_in[11];
    const float* ln2_b  = (const float*)d_in[12];
    const float* fw1    = (const float*)d_in[13];
    const float* fb1    = (const float*)d_in[14];
    const float* fw2    = (const float*)d_in[15];
    const float* fb2    = (const float*)d_in[16];
    const float* fw3    = (const float*)d_in[17];
    const float* fb3    = (const float*)d_in[18];
    const float* raw    = (const float*)d_in[19];
    const float* objw   = (const float*)d_in[20];
    float* out = (float*)d_out;

    // Workspace carve-up (256B aligned)
    char* w = (char*)d_ws;
    auto take = [&](size_t bytes) { char* p = w; w += (bytes + 255) & ~(size_t)255; return p; };
    float*  ctx   = (float*) take((size_t)DDIM * 4);
    __bf16* entb  = (__bf16*)take((size_t)POP * POP * 2);       // 33.5 MB, L2-resident
    __bf16* combT = (__bf16*)take((size_t)DDIM * POP * 2);      // 4 MB
    float*  qe    = (float*) take((size_t)POP * DDIM * 4);      // 8 MB
    float*  fit   = (float*) take((size_t)POP * 4);
    float*  popA  = (float*) take((size_t)POP * DDIM * 4);      // 8 MB
    float*  popB  = (float*) take((size_t)POP * DDIM * 4);      // 8 MB

    enc_kernel<<<1, 512, 0, stream>>>(pc, qkv_w, qkv_b, out_w, out_b,
                                      ff1_w, ff1_b, ff2_w, ff2_b,
                                      ln1_w, ln1_b, ln2_w, ln2_b, ctx);
    softmax_bf16_kernel<<<POP, 256, 0, stream>>>(raw, entb);
    initpop_kernel<<<(POP * DDIM) / 256, 256, 0, stream>>>(ctx, popA);

    float* pop  = popA;
    float* popn = popB;
    for (int gen = 0; gen < GENS; ++gen) {
        combt_kernel<<<(DDIM * POP) / 256, 256, 0, stream>>>(pop, ctx, combT);
        gemm_qe_kernel<<<dim3(DDIM / 128, POP / 128), 256, 0, stream>>>(entb, combT, pop, ctx, qe);
        fitness_kernel<<<POP, 256, 0, stream>>>(qe, fw1, fb1, fw2, fb2, fw3, fb3, fit, gen);
        reproduce_kernel<<<POP / 2, 128, 0, stream>>>(pop, fit, popn, gen);
        float* t = pop; pop = popn; popn = t;
    }
    finalize_kernel<<<1, 512, 0, stream>>>(pop, fit, objw, out);
}
